// Attn_32238024524195
// MI455X (gfx1250) — matrix operand order repaired and checked by reading, not compiled, not run on hardware
//
#include <hip/hip_runtime.h>
#include <math.h>

// ---------------- types / constants ----------------
typedef __attribute__((ext_vector_type(16))) __bf16 v16bf;
typedef __attribute__((ext_vector_type(8)))  __bf16 v8bf;
typedef __attribute__((ext_vector_type(8)))  float  v8f;

static constexpr int Bb  = 2;
static constexpr int Tt  = 2048;
static constexpr int Cc  = 2048;
static constexpr int Hh  = 16;
static constexpr int KVh = 4;
static constexpr int Dd  = 128;
static constexpr int Gg  = Hh / KVh;
static constexpr int BT  = Bb * Tt;

#define DEV __device__ __forceinline__

DEV v16bf combine8(v8bf lo, v8bf hi) {
  v16bf a;
#pragma unroll
  for (int i = 0; i < 8; ++i) { a[i] = lo[i]; a[i + 8] = hi[i]; }
  return a;
}

DEV v8f wmma_bf16(v16bf a, v16bf b, v8f c) {
  // D = A(16x32 bf16) * B(32x16 bf16) + C(16x16 f32)
  return __builtin_amdgcn_wmma_f32_16x16x32_bf16(false, a, false, b, (short)0, c,
                                                 false, false);
}

// A fragment: lane holds M = row; element i -> K = 2*base + i
DEV v16bf loadA(const __bf16* row, int kk, int base) {
  v8bf lo = *(const v8bf*)(row + kk + 2 * base);
  v8bf hi = *(const v8bf*)(row + kk + 2 * base + 8);
  return combine8(lo, hi);
}

// ---------------- elementwise f32 -> bf16 ----------------
__global__ void k_cvt(const float* __restrict__ src, __bf16* __restrict__ dst,
                      size_t n) {
  size_t i = (size_t)blockIdx.x * blockDim.x + threadIdx.x;
  if (i < n) dst[i] = (__bf16)src[i];
}

// ---------------- f32 [K,N] -> bf16 transposed [N,K] ----------------
__global__ void k_cvt_tr(const float* __restrict__ W, __bf16* __restrict__ Wt,
                         int K, int N) {
  size_t i = (size_t)blockIdx.x * blockDim.x + threadIdx.x;
  if (i >= (size_t)K * N) return;
  int n = (int)(i % N);
  int k = (int)(i / N);
  Wt[(size_t)n * K + k] = (__bf16)W[i];
}

// ---------------- RoPE + relayout [B,T,NH,D] -> [B,NH,T,D] ----------------
__global__ void k_rope(const __bf16* __restrict__ in, __bf16* __restrict__ out,
                       int NH) {
  size_t i = (size_t)blockIdx.x * blockDim.x + threadIdx.x;
  size_t total = (size_t)Bb * Tt * NH * (Dd / 2);
  if (i >= total) return;
  int d = (int)(i % (Dd / 2));
  size_t r = i / (Dd / 2);
  int h = (int)(r % NH); r /= NH;
  int t = (int)(r % Tt);
  int b = (int)(r / Tt);
  const __bf16* src = in + (((size_t)b * Tt + t) * NH + h) * Dd;
  float x1 = (float)src[d];
  float x2 = (float)src[d + Dd / 2];
  // inv_freq = 10000^(-2d/D) = exp(-(2d/D)*ln(10000))
  float ang = (float)t * __expf(-(2.0f * (float)d / (float)Dd) * 9.210340371976184f);
  float s, c;
  __sincosf(ang, &s, &c);
  __bf16* dst = out + (((size_t)b * NH + h) * Tt + t) * Dd;
  dst[d]          = (__bf16)(x1 * c - x2 * s);
  dst[d + Dd / 2] = (__bf16)(x2 * c + x1 * s);
}

// ---------------- V relayout [B,T,KV,D] -> [B,KV,D,T] ----------------
__global__ void k_vtr(const __bf16* __restrict__ in, __bf16* __restrict__ out) {
  size_t i = (size_t)blockIdx.x * blockDim.x + threadIdx.x;
  size_t total = (size_t)BT * KVh * Dd;
  if (i >= total) return;
  int d = (int)(i % Dd);
  size_t r = i / Dd;
  int kv = (int)(r % KVh); r /= KVh;
  int t = (int)(r % Tt);
  int b = (int)(r / Tt);
  out[(((size_t)b * KVh + kv) * Dd + d) * Tt + t] = in[i];
}

// ---------------- generic bf16 GEMM: C[M,N] = A[M,K] * Bt[N,K]^T ----------------
// block = 128 threads (4 waves); each wave computes a 32(M) x 64(N) register tile
// (2 A-fragments reused over 4 B-fragments, 8 WMMAs per K-step), with explicit
// double buffering so the next K-step's fragments load under the current WMMAs.
template <bool F32OUT>
__global__ __launch_bounds__(128, 1) void k_gemm(const __bf16* __restrict__ A,
                                                 const __bf16* __restrict__ Bt,
                                                 void* __restrict__ Cout,
                                                 int M, int N, int K) {
  const int lane = threadIdx.x & 31;
  const int wave = threadIdx.x >> 5;
  const int lm   = lane & 15;
  const int half = lane >> 4;
  const int base = half * 8;
  const int m0 = blockIdx.y * 128 + wave * 32;
  const int n0 = blockIdx.x * 64;

  v8f acc[2][4];
#pragma unroll
  for (int mh = 0; mh < 2; ++mh)
#pragma unroll
    for (int t = 0; t < 4; ++t)
#pragma unroll
      for (int r = 0; r < 8; ++r) acc[mh][t][r] = 0.0f;

  const __bf16* arow0 = A + (size_t)(m0 + lm) * K;
  const __bf16* arow1 = A + (size_t)(m0 + 16 + lm) * K;
  const __bf16* brows[4];
#pragma unroll
  for (int t = 0; t < 4; ++t)
    brows[t] = Bt + (size_t)(n0 + t * 16 + lm) * K + half * 16;

  // prologue
  v16bf a0 = loadA(arow0, 0, base);
  v16bf a1 = loadA(arow1, 0, base);
  v16bf bb[4];
#pragma unroll
  for (int t = 0; t < 4; ++t) bb[t] = *(const v16bf*)(brows[t]);

  for (int kk = 0; kk < K; kk += 32) {
    const int kn = kk + 32;
    v16bf na0, na1, nb[4];
    if (kn < K) {
      na0 = loadA(arow0, kn, base);
      na1 = loadA(arow1, kn, base);
#pragma unroll
      for (int t = 0; t < 4; ++t) nb[t] = *(const v16bf*)(brows[t] + kn);
    }
#pragma unroll
    for (int t = 0; t < 4; ++t) {
      acc[0][t] = wmma_bf16(a0, bb[t], acc[0][t]);
      acc[1][t] = wmma_bf16(a1, bb[t], acc[1][t]);
    }
    if (kn < K) {
      a0 = na0;
      a1 = na1;
#pragma unroll
      for (int t = 0; t < 4; ++t) bb[t] = nb[t];
    }
  }

#pragma unroll
  for (int mh = 0; mh < 2; ++mh)
#pragma unroll
    for (int t = 0; t < 4; ++t)
#pragma unroll
      for (int r = 0; r < 8; ++r) {
        size_t row = (size_t)m0 + mh * 16 + r + 8 * half;
        size_t col = (size_t)n0 + t * 16 + lm;
        if (F32OUT)
          ((float*)Cout)[row * N + col] = acc[mh][t][r];
        else
          ((__bf16*)Cout)[row * N + col] = (__bf16)acc[mh][t][r];
      }
}

// ---------------- flash attention ----------------
// grid = (T/64, B*H); block = 128 (4 waves); wave owns one 16-row Q tile.
// Q: [B,H,T,D] bf16 (post-RoPE); K: [B,KV,T,D] bf16 (post-RoPE); V: [B,KV,D,T] bf16.
// Online softmax stats (m, l, alpha) live in registers: WMMA C-layout row
// mapping (row = r + 8*half) matches the O accumulators, so rescale/normalize
// are pure register ops; row max/sum use xor-shuffles within each 16-lane half.
// LDS is used only for the C-layout -> A-fragment repack of P.
__global__ __launch_bounds__(128, 1) void k_attn(const __bf16* __restrict__ Qr,
                                                 const __bf16* __restrict__ Kr,
                                                 const __bf16* __restrict__ Vt,
                                                 __bf16* __restrict__ Obf) {
  __shared__ __align__(32) __bf16 Pbuf[4][16][32];

  const int lane = threadIdx.x & 31;
  const int wave = threadIdx.x >> 5;
  const int lm   = lane & 15;
  const int half = lane >> 4;
  const int base = half * 8;

  const int bh  = blockIdx.y;
  const int b   = bh / Hh;
  const int h   = bh % Hh;
  const int kvh = h / Gg;
  const int q0  = (blockIdx.x * 4 + wave) * 16;

  const __bf16* Q  = Qr + ((size_t)(b * Hh + h) * Tt) * Dd;
  const __bf16* Kc = Kr + ((size_t)(b * KVh + kvh) * Tt) * Dd;
  const __bf16* Vh = Vt + ((size_t)(b * KVh + kvh) * Dd) * Tt;

  // Preload Q A-fragments for the 4 K-dim chunks (D=128 = 4 x 32)
  v16bf qa[4];
  const __bf16* qrow = Q + (size_t)(q0 + lm) * Dd;
#pragma unroll
  for (int kc = 0; kc < 4; ++kc) qa[kc] = loadA(qrow, kc * 32, base);

  float m_r[8], l_r[8];
#pragma unroll
  for (int r = 0; r < 8; ++r) { m_r[r] = -INFINITY; l_r[r] = 0.0f; }
  v8f o[8];
#pragma unroll
  for (int dt = 0; dt < 8; ++dt)
#pragma unroll
    for (int r = 0; r < 8; ++r) o[dt][r] = 0.0f;

  const float scale = 0.08838834764831845f;  // 1/sqrt(128)
  const int ntiles  = (q0 + 16 + 31) >> 5;   // causal: kv tiles of 32 tokens

  for (int kt = 0; kt < ntiles; ++kt) {
    const int t0 = kt * 32;

    // prefetch next K tile rows (global_prefetch path; K stream is predictable)
    if (kt + 1 < ntiles) {
      __builtin_prefetch((const void*)(Kc + (size_t)(t0 + 32 + lm) * Dd), 0, 1);
      __builtin_prefetch((const void*)(Kc + (size_t)(t0 + 48 + lm) * Dd), 0, 1);
    }

    // ---- S = Q * K^T (16 x 32 tile = two 16x16 WMMA outputs) ----
    v8f s0, s1;
#pragma unroll
    for (int r = 0; r < 8; ++r) { s0[r] = 0.0f; s1[r] = 0.0f; }
#pragma unroll
    for (int kc = 0; kc < 4; ++kc) {
      v16bf kb0 = *(const v16bf*)(Kc + (size_t)(t0 + lm) * Dd + kc * 32 + half * 16);
      v16bf kb1 = *(const v16bf*)(Kc + (size_t)(t0 + 16 + lm) * Dd + kc * 32 + half * 16);
      s0 = wmma_bf16(qa[kc], kb0, s0);
      s1 = wmma_bf16(qa[kc], kb1, s1);
    }

    // ---- register-resident online softmax ----
    // Lane holds cols (t0+lm, t0+16+lm) of rows r+8*half; reduce over the
    // 16 lanes of the half with xor-shuffles (masks 1..8 stay in-half).
    float alpha[8];
#pragma unroll
    for (int r = 0; r < 8; ++r) {
      const int rowg = q0 + r + 8 * half;               // global query row
      const bool ok0 = (t0 + lm) <= rowg;
      const bool ok1 = (t0 + 16 + lm) <= rowg;
      float v0 = ok0 ? s0[r] * scale : -INFINITY;
      float v1 = ok1 ? s1[r] * scale : -INFINITY;
      float mx = fmaxf(v0, v1);
#pragma unroll
      for (int d = 1; d < 16; d <<= 1) mx = fmaxf(mx, __shfl_xor(mx, d, 32));
      const float mnew = fmaxf(m_r[r], mx);
      float p0 = ok0 ? __expf(v0 - mnew) : 0.0f;
      float p1 = ok1 ? __expf(v1 - mnew) : 0.0f;
      float rs = p0 + p1;
#pragma unroll
      for (int d = 1; d < 16; d <<= 1) rs += __shfl_xor(rs, d, 32);
      alpha[r] = __expf(m_r[r] - mnew);
      m_r[r]   = mnew;
      l_r[r]   = l_r[r] * alpha[r] + rs;
      // C-layout -> row-major P in LDS (bf16), ready for A-fragment reload
      Pbuf[wave][r + 8 * half][lm]      = (__bf16)p0;
      Pbuf[wave][r + 8 * half][16 + lm] = (__bf16)p1;
    }

    // rescale O accumulators (same row mapping as alpha -> pure register op)
#pragma unroll
    for (int dt = 0; dt < 8; ++dt)
#pragma unroll
      for (int r = 0; r < 8; ++r) o[dt][r] *= alpha[r];

    // ---- O += P * V (V fragments loaded in pairs to overlap with WMMA) ----
    const __bf16* prow = &Pbuf[wave][lm][0];
    v8bf plo = *(const v8bf*)(prow + 2 * base);
    v8bf phi = *(const v8bf*)(prow + 2 * base + 8);
    v16bf pa = combine8(plo, phi);
#pragma unroll
    for (int dt = 0; dt < 8; dt += 2) {
      v16bf vb0 = *(const v16bf*)(Vh + (size_t)(dt * 16 + lm) * Tt + t0 + half * 16);
      v16bf vb1 = *(const v16bf*)(Vh + (size_t)((dt + 1) * 16 + lm) * Tt + t0 + half * 16);
      o[dt]     = wmma_bf16(pa, vb0, o[dt]);
      o[dt + 1] = wmma_bf16(pa, vb1, o[dt + 1]);
    }
  }

  // ---- normalize (register-resident l) and store to [B,T,H*D] bf16 ----
  float invl[8];
#pragma unroll
  for (int r = 0; r < 8; ++r) invl[r] = 1.0f / l_r[r];
#pragma unroll
  for (int dt = 0; dt < 8; ++dt)
#pragma unroll
    for (int r = 0; r < 8; ++r) {
      size_t row = (size_t)b * Tt + q0 + r + 8 * half;
      size_t col = (size_t)h * Dd + dt * 16 + lm;
      Obf[row * Cc + col] = (__bf16)(o[dt][r] * invl[r]);
    }
}

// ---------------- host orchestration ----------------
extern "C" void kernel_launch(void* const* d_in, const int* in_sizes, int n_in,
                              void* d_out, int out_size, void* d_ws, size_t ws_size,
                              hipStream_t stream) {
  const float* x  = (const float*)d_in[0];
  const float* Wq = (const float*)d_in[1];
  const float* Wk = (const float*)d_in[2];
  const float* Wv = (const float*)d_in[3];
  const float* Wo = (const float*)d_in[4];
  float* out = (float*)d_out;

  char* ws = (char*)d_ws;
  size_t off = 0;
  auto carve = [&](size_t bytes) -> void* {
    off = (off + 255) & ~(size_t)255;
    void* p = ws + off;
    off += bytes;
    return p;
  };

  __bf16* xb    = (__bf16*)carve((size_t)BT * Cc * 2);          // x bf16 [B*T, C]
  __bf16* Wqt   = (__bf16*)carve((size_t)Cc * (Hh * Dd) * 2);   // [N=2048][K=2048]
  __bf16* Wkt   = (__bf16*)carve((size_t)(KVh * Dd) * Cc * 2);  // [512][2048]
  __bf16* Wvt   = (__bf16*)carve((size_t)(KVh * Dd) * Cc * 2);
  __bf16* Wot   = (__bf16*)carve((size_t)Cc * Cc * 2);
  __bf16* q_raw = (__bf16*)carve((size_t)BT * (Hh * Dd) * 2);   // [B,T,H,D]
  __bf16* k_raw = (__bf16*)carve((size_t)BT * (KVh * Dd) * 2);  // [B,T,KV,D]
  __bf16* v_raw = (__bf16*)carve((size_t)BT * (KVh * Dd) * 2);
  __bf16* q_r   = (__bf16*)carve((size_t)BT * (Hh * Dd) * 2);   // [B,H,T,D]
  __bf16* k_r   = (__bf16*)carve((size_t)BT * (KVh * Dd) * 2);  // [B,KV,T,D]
  __bf16* v_t   = (__bf16*)carve((size_t)BT * (KVh * Dd) * 2);  // [B,KV,D,T]
  __bf16* o_bf  = q_raw;  // reuse: q_raw dead after RoPE; same size as O [B,T,C]

  const int TB = 256;
  // convert activations + weights (transpose weights into [N][K])
  {
    size_t n = (size_t)BT * Cc;
    k_cvt<<<dim3((unsigned)((n + TB - 1) / TB)), TB, 0, stream>>>(x, xb, n);
  }
  k_cvt_tr<<<dim3((unsigned)(((size_t)Cc * 2048 + TB - 1) / TB)), TB, 0, stream>>>(Wq, Wqt, Cc, 2048);
  k_cvt_tr<<<dim3((unsigned)(((size_t)Cc * 512  + TB - 1) / TB)), TB, 0, stream>>>(Wk, Wkt, Cc, 512);
  k_cvt_tr<<<dim3((unsigned)(((size_t)Cc * 512  + TB - 1) / TB)), TB, 0, stream>>>(Wv, Wvt, Cc, 512);
  k_cvt_tr<<<dim3((unsigned)(((size_t)Cc * 2048 + TB - 1) / TB)), TB, 0, stream>>>(Wo, Wot, Cc, 2048);

  // QKV projections (bf16 WMMA, f32 accumulate, bf16 out)
  k_gemm<false><<<dim3(2048 / 64, BT / 128), 128, 0, stream>>>(xb, Wqt, q_raw, BT, 2048, Cc);
  k_gemm<false><<<dim3(512  / 64, BT / 128), 128, 0, stream>>>(xb, Wkt, k_raw, BT, 512, Cc);
  k_gemm<false><<<dim3(512  / 64, BT / 128), 128, 0, stream>>>(xb, Wvt, v_raw, BT, 512, Cc);

  // RoPE + relayout
  {
    size_t nq = (size_t)Bb * Tt * Hh * (Dd / 2);
    k_rope<<<dim3((unsigned)((nq + TB - 1) / TB)), TB, 0, stream>>>(q_raw, q_r, Hh);
    size_t nk = (size_t)Bb * Tt * KVh * (Dd / 2);
    k_rope<<<dim3((unsigned)((nk + TB - 1) / TB)), TB, 0, stream>>>(k_raw, k_r, KVh);
    size_t nv = (size_t)BT * KVh * Dd;
    k_vtr<<<dim3((unsigned)((nv + TB - 1) / TB)), TB, 0, stream>>>(v_raw, v_t);
  }

  // causal flash attention
  k_attn<<<dim3(Tt / 64, Bb * Hh), 128, 0, stream>>>(q_r, k_r, v_t, o_bf);

  // output projection -> f32 result
  k_gemm<true><<<dim3(Cc / 64, BT / 128), 128, 0, stream>>>(o_bf, Wot, out, BT, Cc, Cc);
}
